// TriangleAttention_7026566497089
// MI455X (gfx1250) — matrix operand order repaired and checked by reading, not compiled, not run on hardware
//
#include <hip/hip_runtime.h>
#include <hip/hip_bf16.h>
#include <math.h>

// ---------------------------------------------------------------------------
// TriangleAttention for MI455X (gfx1250, wave32, WMMA bf16 w/ f32 accum)
// B=1, N=256, DIM=128, H=4, hd=32
// ---------------------------------------------------------------------------

typedef __attribute__((ext_vector_type(16))) __bf16  v16bf;
typedef __attribute__((ext_vector_type(2)))  __bf16  v2bf;
typedef __attribute__((ext_vector_type(8)))  float   v8f;
typedef __attribute__((ext_vector_type(8)))  unsigned int v8u;

#define NTOK   256
#define DIM    128
#define NHEAD  4
#define HD     32
#define SCALE  0.08838834764831845f   // 128^-0.5

// Native f32 -> bf16 (backend selects hardware cvt on gfx1250)
__device__ __forceinline__ unsigned short f2bf(float x) {
    return __builtin_bit_cast(unsigned short, (__bf16)x);
}

// Pack two f32 -> one dword of 2x bf16; prefer the packed hardware cvt.
__device__ __forceinline__ unsigned int pack2(float a, float b) {
#if __has_builtin(__builtin_amdgcn_cvt_pk_bf16_f32)
    v2bf t = __builtin_amdgcn_cvt_pk_bf16_f32(a, b);
    return __builtin_bit_cast(unsigned int, t);
#else
    unsigned int ua = __builtin_bit_cast(unsigned int, a);
    unsigned int ub = __builtin_bit_cast(unsigned int, b);
    ua = (ua + 0x7FFFu + ((ua >> 16) & 1u)) >> 16;
    ub = (ub + 0x7FFFu + ((ub >> 16) & 1u)) >> 16;
    return ua | (ub << 16);
#endif
}

__device__ __forceinline__ v16bf as_bf(v8u x) { return __builtin_bit_cast(v16bf, x); }

// A-fragment (16x32 bf16), source row-major [row][k], rowStride in elements.
// ISA layout: lane m = lane&15; half hh = lane>>4;
//   VGPR j<4 : K = hh*8 + 2j, +1 ; VGPR j>=4 : K = 16 + hh*8 + 2(j-4), +1
__device__ __forceinline__ v8u fragA(const unsigned short* base, int row0, int col0,
                                     int rowStride, int lane) {
    int m = lane & 15, hh = lane >> 4;
    v8u f;
#pragma unroll
    for (int j = 0; j < 8; ++j) {
        int k0 = ((j >> 2) << 4) + hh * 8 + ((j & 3) << 1);
        f[j] = *(const unsigned int*)(base + (size_t)(row0 + m) * rowStride + col0 + k0);
    }
    return f;
}

// B-fragment (32x16 bf16), source layout: column n has K contiguous,
// addr = (col0+n)*colStride + k0base + k ; lane n = lane&15, hh = lane>>4,
// VGPR j<4 : K = hh*8 + 2j, +1 ; VGPR j>=4 : K = 16 + hh*8 + 2(j-4), +1
__device__ __forceinline__ v8u fragB(const unsigned short* base, int col0, int k0base,
                                     int colStride, int lane) {
    int n = lane & 15, hh = lane >> 4;
    v8u f;
#pragma unroll
    for (int j = 0; j < 8; ++j) {
        int k0 = ((j >> 2) << 4) + hh * 8 + ((j & 3) << 1);
        f[j] = *(const unsigned int*)(base + (size_t)(col0 + n) * colStride + k0base + k0);
    }
    return f;
}

__device__ __forceinline__ v8f wmma_bf16(v8u a, v8u b, v8f c) {
    return __builtin_amdgcn_wmma_f32_16x16x32_bf16(
        false, as_bf(a), false, as_bf(b), (short)0, c, false, false);
}

// ---------------------------------------------------------------------------
// Kernel 0: weights f32 [c][d] -> bf16 transposed [d][c] (5 matrices packed)
// ---------------------------------------------------------------------------
__global__ void k_weights(const float* __restrict__ Wq, const float* __restrict__ Wk,
                          const float* __restrict__ Wv, const float* __restrict__ Wo,
                          const float* __restrict__ Wg, unsigned short* __restrict__ WT) {
    int idx = blockIdx.x * 256 + threadIdx.x;        // < 5*16384
    int w = idx >> 14;
    int rem = idx & 16383;
    int d = rem >> 7, c = rem & 127;
    const float* src = (w == 0) ? Wq : (w == 1) ? Wk : (w == 2) ? Wv : (w == 3) ? Wo : Wg;
    WT[idx] = f2bf(src[c * DIM + d]);
}

// ---------------------------------------------------------------------------
// Kernel 1: biasT[h][q][k] = sum_c pair[q][k][c] * Wb[c][h]   (f32)
// ---------------------------------------------------------------------------
__global__ void k_bias(const float* __restrict__ pair, const float* __restrict__ Wb,
                       float* __restrict__ biasT) {
    int idx = blockIdx.x * 256 + threadIdx.x;        // (q,k) flat, < 65536
    const float4* p4 = (const float4*)(pair + (size_t)idx * DIM);
    float acc0 = 0.f, acc1 = 0.f, acc2 = 0.f, acc3 = 0.f;
#pragma unroll 8
    for (int c4 = 0; c4 < DIM / 4; ++c4) {
        float4 p = p4[c4];
        int c = c4 * 4;
        acc0 += p.x * Wb[(c + 0) * NHEAD + 0] + p.y * Wb[(c + 1) * NHEAD + 0] +
                p.z * Wb[(c + 2) * NHEAD + 0] + p.w * Wb[(c + 3) * NHEAD + 0];
        acc1 += p.x * Wb[(c + 0) * NHEAD + 1] + p.y * Wb[(c + 1) * NHEAD + 1] +
                p.z * Wb[(c + 2) * NHEAD + 1] + p.w * Wb[(c + 3) * NHEAD + 1];
        acc2 += p.x * Wb[(c + 0) * NHEAD + 2] + p.y * Wb[(c + 1) * NHEAD + 2] +
                p.z * Wb[(c + 2) * NHEAD + 2] + p.w * Wb[(c + 3) * NHEAD + 2];
        acc3 += p.x * Wb[(c + 0) * NHEAD + 3] + p.y * Wb[(c + 1) * NHEAD + 3] +
                p.z * Wb[(c + 2) * NHEAD + 3] + p.w * Wb[(c + 3) * NHEAD + 3];
    }
    biasT[0 * 65536 + idx] = acc0;
    biasT[1 * 65536 + idx] = acc1;
    biasT[2 * 65536 + idx] = acc2;
    biasT[3 * 65536 + idx] = acc3;
}

// ---------------------------------------------------------------------------
// Kernel 2: per-row fused QKV projection + attention.
// grid = 256 (row i), block = 256 (8 wave32), dyn LDS = 256KB:
//   [  0KB) pair bf16 [256][128]  (phase1/2) -> reused as per-wave P scratch
//   [ 64KB) q    bf16 [256][128]  row-major
//   [128KB) k    bf16 [256][128]  row-major  (B-frags: col=key, K=dim contig)
//   [192KB) v    bf16 [128][256]  transposed (B-frags: col=dim, K=key contig)
// ---------------------------------------------------------------------------
__global__ __launch_bounds__(256) void k_attn(const float* __restrict__ pair,
                                              const unsigned short* __restrict__ WT, // 3x [d][c]
                                              const float* __restrict__ biasT,
                                              unsigned short* __restrict__ attn) {
    extern __shared__ unsigned char smem[];
    unsigned short* pair_b = (unsigned short*)smem;               // 32768 elems
    unsigned short* q_b    = (unsigned short*)(smem + 65536);
    unsigned short* k_b    = (unsigned short*)(smem + 131072);
    unsigned short* v_b    = (unsigned short*)(smem + 196608);

    const int tid  = threadIdx.x;
    const int lane = tid & 31;
    const int wave = tid >> 5;
    const int i    = blockIdx.x;
    const int nl   = lane & 15;
    const int mb8  = (lane >> 4) * 8;

    // ---- Phase 1: stage pair row i as bf16 ----
    const float2* src2 = (const float2*)(pair + (size_t)i * NTOK * DIM);
    unsigned int* pb32 = (unsigned int*)pair_b;
#pragma unroll 4
    for (int idx = tid; idx < NTOK * DIM / 2; idx += 256) {
        float2 p = src2[idx];
        pb32[idx] = pack2(p.x, p.y);
    }
    __syncthreads();

    // ---- Phase 2: q,k,v = pair_row @ W{q,k,v}  (384 16x16 tiles / 8 waves) ----
    for (int t = wave; t < 384; t += 8) {
        int mat = t >> 7;                 // 0=q 1=k 2=v
        int rem = t & 127;
        int mb = rem >> 3, nb = rem & 7;
        const unsigned short* W = WT + mat * 16384;
        v8f acc = {0.f, 0.f, 0.f, 0.f, 0.f, 0.f, 0.f, 0.f};
#pragma unroll
        for (int ks = 0; ks < 4; ++ks) {
            v8u a = fragA(pair_b, mb * 16, ks * 32, DIM, lane);
            v8u b = fragB(W, nb * 16, ks * 32, DIM, lane);   // global (L2-resident)
            acc = wmma_bf16(a, b, acc);
        }
        if (mat == 2) {
            // v transposed: [dim][key]; r,r+1 are adjacent keys -> packed b32 stores
            unsigned int* vrow =
                (unsigned int*)(v_b + (size_t)(nb * 16 + nl) * NTOK + mb * 16 + mb8);
#pragma unroll
            for (int r = 0; r < 4; ++r)
                vrow[r] = pack2(acc[2 * r], acc[2 * r + 1]);
        } else {
            unsigned short* dst = (mat == 0) ? q_b : k_b;
#pragma unroll
            for (int r = 0; r < 8; ++r)
                dst[(size_t)(mb * 16 + mb8 + r) * DIM + nb * 16 + nl] = f2bf(acc[r]);
        }
    }
    __syncthreads();   // pair_b now free -> per-wave P scratch

    unsigned short* psc = pair_b + wave * (16 * NTOK);   // 8KB per wave

    // ---- Phase 3: attention, 64 units (16 qblocks x 4 heads) / 8 waves ----
    for (int u = wave; u < 64; u += 8) {
        int qb = u >> 2, h = u & 3;

        v8u qf = fragA(q_b, qb * 16, h * HD, DIM, lane);

        float sc[16][8];
        float pm[8];
#pragma unroll
        for (int r = 0; r < 8; ++r) pm[r] = -1e30f;

        const float* bT = biasT + (size_t)h * 65536 + (size_t)(qb * 16 + mb8) * NTOK;

        // scores: 16 key-blocks of 16, one WMMA each (K = hd = 32)
#pragma unroll
        for (int kb = 0; kb < 16; ++kb) {
            v8u kf = fragB(k_b, kb * 16, h * HD, DIM, lane);
            v8f z = {0.f, 0.f, 0.f, 0.f, 0.f, 0.f, 0.f, 0.f};
            v8f a = wmma_bf16(qf, kf, z);
#pragma unroll
            for (int r = 0; r < 8; ++r) {
                float s = a[r] * SCALE + bT[(size_t)r * NTOK + kb * 16 + nl];
                sc[kb][r] = s;
                pm[r] = fmaxf(pm[r], s);
            }
        }
        // row-max across the 16-lane half (rows live at fixed (half, r))
#pragma unroll
        for (int r = 0; r < 8; ++r) {
#pragma unroll
            for (int d = 1; d < 16; d <<= 1)
                pm[r] = fmaxf(pm[r], __shfl_xor(pm[r], d, 32));
        }
        float es[8];
#pragma unroll
        for (int r = 0; r < 8; ++r) es[r] = 0.f;
#pragma unroll
        for (int kb = 0; kb < 16; ++kb) {
#pragma unroll
            for (int r = 0; r < 8; ++r) {
                float e = __expf(sc[kb][r] - pm[r]);
                es[r] += e;
                psc[(size_t)(mb8 + r) * NTOK + kb * 16 + nl] = f2bf(e);
            }
        }
#pragma unroll
        for (int r = 0; r < 8; ++r) {
#pragma unroll
            for (int d = 1; d < 16; d <<= 1)
                es[r] += __shfl_xor(es[r], d, 32);
        }
        float rinv[8];
#pragma unroll
        for (int r = 0; r < 8; ++r) rinv[r] = 1.0f / es[r];

        // out = P(16x256) @ V(256x32): 8 K-steps x 2 column tiles
        v8f o0 = {0.f, 0.f, 0.f, 0.f, 0.f, 0.f, 0.f, 0.f};
        v8f o1 = {0.f, 0.f, 0.f, 0.f, 0.f, 0.f, 0.f, 0.f};
#pragma unroll
        for (int kc = 0; kc < 8; ++kc) {
            v8u pa = fragA(psc, 0, kc * 32, NTOK, lane);
            v8u b0 = fragB(v_b, h * HD, kc * 32, NTOK, lane);
            v8u b1 = fragB(v_b, h * HD + 16, kc * 32, NTOK, lane);
            o0 = wmma_bf16(pa, b0, o0);
            o1 = wmma_bf16(pa, b1, o1);
        }
        size_t rowbase = (size_t)i * NTOK + qb * 16 + mb8;
#pragma unroll
        for (int r = 0; r < 8; ++r) {
            size_t row = rowbase + r;
            attn[row * DIM + h * HD + nl]      = f2bf(o0[r] * rinv[r]);
            attn[row * DIM + h * HD + 16 + nl] = f2bf(o1[r] * rinv[r]);
        }
    }
}

// ---------------------------------------------------------------------------
// Kernel 3: out = (attn @ Wo) * sigmoid(pair @ Wg)   (64-row tiles)
// dyn LDS 32KB: attn_s bf16 [64][128] ; pair_s bf16 [64][128]
// ---------------------------------------------------------------------------
__global__ __launch_bounds__(256) void k_out(const unsigned short* __restrict__ attn,
                                             const float* __restrict__ pair,
                                             const unsigned short* __restrict__ WoT,
                                             const unsigned short* __restrict__ WgT,
                                             float* __restrict__ out) {
    extern __shared__ unsigned char smem[];
    unsigned short* attn_s = (unsigned short*)smem;              // 8192 elems
    unsigned short* pair_s = (unsigned short*)(smem + 16384);

    const int tid  = threadIdx.x;
    const int lane = tid & 31;
    const int wave = tid >> 5;
    const int nl   = lane & 15;
    const int mb8  = (lane >> 4) * 8;
    const size_t rows0 = (size_t)blockIdx.x * 64;

    const unsigned int* asrc = (const unsigned int*)(attn + rows0 * DIM);
    const float2* psrc = (const float2*)(pair + rows0 * DIM);
    unsigned int* ad = (unsigned int*)attn_s;
    unsigned int* pd = (unsigned int*)pair_s;
#pragma unroll 4
    for (int idx = tid; idx < 64 * DIM / 2; idx += 256) {
        ad[idx] = asrc[idx];
        float2 p = psrc[idx];
        pd[idx] = pack2(p.x, p.y);
    }
    __syncthreads();

    for (int t = wave; t < 32; t += 8) {
        int mb = t >> 3, nb = t & 7;
        v8f accO = {0.f, 0.f, 0.f, 0.f, 0.f, 0.f, 0.f, 0.f};
        v8f accG = {0.f, 0.f, 0.f, 0.f, 0.f, 0.f, 0.f, 0.f};
#pragma unroll
        for (int ks = 0; ks < 4; ++ks) {
            v8u aO = fragA(attn_s, mb * 16, ks * 32, DIM, lane);
            v8u bO = fragB(WoT, nb * 16, ks * 32, DIM, lane);
            accO = wmma_bf16(aO, bO, accO);
            v8u aG = fragA(pair_s, mb * 16, ks * 32, DIM, lane);
            v8u bG = fragB(WgT, nb * 16, ks * 32, DIM, lane);
            accG = wmma_bf16(aG, bG, accG);
        }
#pragma unroll
        for (int r = 0; r < 8; ++r) {
            size_t row = rows0 + mb * 16 + mb8 + r;
            float g = 1.0f / (1.0f + __expf(-accG[r]));
            out[row * DIM + nb * 16 + nl] = accO[r] * g;
        }
    }
}

// ---------------------------------------------------------------------------
// Host launcher
// ws layout (bytes):
//   [0)        WqT/WkT/WvT/WoT/WgT bf16, 5 * 32768 = 163840
//   [163840)   biasT f32 [4][256][256]   = 1048576
//   [1212416)  attn bf16 [65536][128]    = 16777216
// total ~17.2 MB
// ---------------------------------------------------------------------------
extern "C" void kernel_launch(void* const* d_in, const int* in_sizes, int n_in,
                              void* d_out, int out_size, void* d_ws, size_t ws_size,
                              hipStream_t stream) {
    const float* pair = (const float*)d_in[0];
    const float* Wq   = (const float*)d_in[1];
    const float* Wk   = (const float*)d_in[2];
    const float* Wv   = (const float*)d_in[3];
    const float* Wb   = (const float*)d_in[4];
    const float* Wo   = (const float*)d_in[5];
    const float* Wg   = (const float*)d_in[6];

    unsigned char* ws = (unsigned char*)d_ws;
    unsigned short* WT    = (unsigned short*)ws;               // 5 packed
    unsigned short* WqkvT = WT;                                // mats 0..2
    unsigned short* WoT   = WT + 3 * 16384;
    unsigned short* WgT   = WT + 4 * 16384;
    float*          biasT = (float*)(ws + 163840);
    unsigned short* attn  = (unsigned short*)(ws + 163840 + 1048576);

    k_weights<<<320, 256, 0, stream>>>(Wq, Wk, Wv, Wo, Wg, WT);
    k_bias<<<256, 256, 0, stream>>>(pair, Wb, biasT);
    k_attn<<<256, 256, 262144, stream>>>(pair, WqkvT, biasT, attn);
    k_out<<<1024, 256, 32768, stream>>>(attn, pair, WoT, WgT, (float*)d_out);
}
